// TransformerDecoderLayers_25340307046453
// MI455X (gfx1250) — matrix operand order repaired and checked by reading, not compiled, not run on hardware
//
#include <hip/hip_runtime.h>
#include <hip/hip_bf16.h>
#include <stdint.h>

typedef __bf16 bf16_t;
typedef __attribute__((ext_vector_type(16))) __bf16 v16bf;
typedef __attribute__((ext_vector_type(8)))  float  v8f;

#define DEV static __device__ __forceinline__

DEV bf16_t f2bf(float f) { return (bf16_t)f; }

// Async DMA of 16 bytes: global -> LDS. Tracked by ASYNCcnt.
DEV void async_b128(const void* gaddr, uint32_t lds_byte_off) {
  asm volatile("global_load_async_to_lds_b128 %0, %1, off"
               :: "v"(lds_byte_off), "v"((unsigned long long)(uintptr_t)gaddr)
               : "memory");
}
DEV void wait_async0() { asm volatile("s_wait_asynccnt 0" ::: "memory"); }
DEV uint32_t lds_off(const void* p) { return (uint32_t)(uintptr_t)p; }

// A-matrix fragment (16x32 bf16), ISA 7.12.2 layout:
// lanes 0-15 hold K = {0..7, 16..23}, lanes 16-31 hold K = {8..15, 24..31}
DEV v16bf load_frag_a(const bf16_t* base, int ld, int row0, int k0) {
  const int lane = threadIdx.x & 31;
  const int m = lane & 15, hi = lane >> 4;
  const bf16_t* p = base + (size_t)(row0 + m) * ld + k0 + hi * 8;
  union U { uint4 q; bf16_t h[8]; };
  U a, b;
  a.q = *(const uint4*)(p);
  b.q = *(const uint4*)(p + 16);
  v16bf v;
#pragma unroll
  for (int i = 0; i < 8; ++i) { v[i] = a.h[i]; v[8 + i] = b.h[i]; }
  return v;
}

// B-matrix fragment (32x16 bf16): lane n holds column n of B; lanes 0-15 hold
// K={0..7, 16..23}, lanes 16-31 hold K={8..15, 24..31}. base is row-major [N][K] (B transposed).
DEV v16bf load_frag_b(const bf16_t* base, int ld, int row0, int k0) {
  const int lane = threadIdx.x & 31;
  const int n = lane & 15, hi = lane >> 4;
  const bf16_t* p = base + (size_t)(row0 + n) * ld + k0 + hi * 8;
  union U { uint4 q; bf16_t h[8]; };
  U a, b;
  a.q = *(const uint4*)(p);
  b.q = *(const uint4*)(p + 16);
  v16bf v;
#pragma unroll
  for (int i = 0; i < 8; ++i) { v[i] = a.h[i]; v[8 + i] = b.h[i]; }
  return v;
}

DEV v8f wmma_bf16(v16bf a, v16bf b, v8f c) {
  return __builtin_amdgcn_wmma_f32_16x16x32_bf16(false, a, false, b, (short)0, c,
                                                 false, false);
}

// C[M][N] = A[M][K] @ Wt[N][K]^T + bias (+ optional ReLU). 8 waves/block, each
// wave owns a 16x64 strip. The 64x32 B tile (shared by all 8 waves) is staged
// in LDS via async-DMA, double-buffered so the k+1 tile streams in while k is
// consumed by the WMMA pipe.
__global__ void __launch_bounds__(256)
gemm_bf16_kernel(const bf16_t* __restrict__ A, const bf16_t* __restrict__ Wt,
                 const float* __restrict__ bias, float* __restrict__ Cf,
                 bf16_t* __restrict__ Cb, int M, int N, int K, int relu) {
  __shared__ bf16_t Bt[2][64][32];  // [buf][col-in-tile][k] 4KB each
  const int w = threadIdx.x >> 5;
  const int row0 = blockIdx.x * 128 + w * 16;
  const int col0 = blockIdx.y * 64;
  const int t = threadIdx.x;
  const int brow = t >> 2, bch = t & 3;  // 64 rows x 4 chunks of 8 bf16

  auto issue = [&](int buf, int k0) {
    const bf16_t* g = Wt + (size_t)(col0 + brow) * K + k0 + bch * 8;
    async_b128(g, lds_off(&Bt[buf][brow][bch * 8]));
  };

  issue(0, 0);
  int buf = 0;
  v8f acc0 = {}, acc1 = {}, acc2 = {}, acc3 = {};
  for (int k0 = 0; k0 < K; k0 += 32) {
    wait_async0();
    __syncthreads();
    if (k0 + 32 < K) {
      issue(buf ^ 1, k0 + 32);
      __builtin_prefetch(A + (size_t)row0 * K + k0 + 32, 0, 0);
    }
    v16bf a = load_frag_a(A, K, row0, k0);
    const bf16_t* bb = &Bt[buf][0][0];
    acc0 = wmma_bf16(a, load_frag_b(bb, 32, 0, 0), acc0);
    acc1 = wmma_bf16(a, load_frag_b(bb, 32, 16, 0), acc1);
    acc2 = wmma_bf16(a, load_frag_b(bb, 32, 32, 0), acc2);
    acc3 = wmma_bf16(a, load_frag_b(bb, 32, 48, 0), acc3);
    __syncthreads();
    buf ^= 1;
  }
  const int lane = threadIdx.x & 31;
  const int cn = lane & 15, mh = (lane >> 4) * 8;
  v8f accs[4] = {acc0, acc1, acc2, acc3};
#pragma unroll
  for (int tt = 0; tt < 4; ++tt) {
#pragma unroll
    for (int r = 0; r < 8; ++r) {
      const int mm = row0 + mh + r;
      const int nn = col0 + tt * 16 + cn;
      float v = accs[tt][r] + bias[nn];
      if (relu) v = fmaxf(v, 0.f);
      const size_t idx = (size_t)mm * N + nn;
      if (Cf) Cf[idx] = v;
      if (Cb) Cb[idx] = f2bf(v);
    }
  }
}

// Flash attention: one wave per (head, 16-row Q block); a block's 8 waves share
// one head, so the 32x64 K tile and 64x32 V^T tile per j-step are staged in LDS
// via async-DMA (double buffered) and shared. P (C-layout) re-stripes to
// A-layout through a per-wave LDS tile. No 1/sqrt(d) scale (reference has none).
__global__ void __launch_bounds__(256)
attn_kernel(const bf16_t* __restrict__ Qb, const bf16_t* __restrict__ Kb,
            const bf16_t* __restrict__ Vt, const float* __restrict__ mask,
            int mstride, bf16_t* __restrict__ Ob, int Tq, int Tk) {
  __shared__ bf16_t Kt[2][32][64];   // [buf][k-row][d]      4KB each
  __shared__ bf16_t Vtl[2][64][32];  // [buf][d][k-col]      4KB each
  __shared__ bf16_t Pt[8][16][32];   // per-wave P re-stripe 8KB
  const int w = threadIdx.x >> 5;
  const int head = blockIdx.y;
  const int qrow0 = blockIdx.x * 128 + w * 16;
  const int h64 = head * 64;
  const int lane = threadIdx.x & 31;
  const int cn = lane & 15, hi = lane >> 4;
  const int t = threadIdx.x;
  const int kr = t >> 3, kch = t & 7;  // K tile: 32 rows x 8 chunks
  const int vr = t >> 2, vch = t & 3;  // V tile: 64 rows x 4 chunks

  auto issueK = [&](int buf, int j) {
    const bf16_t* g = Kb + (size_t)(j + kr) * 512 + h64 + kch * 8;
    async_b128(g, lds_off(&Kt[buf][kr][kch * 8]));
  };
  auto issueV = [&](int buf, int j) {
    const bf16_t* g = Vt + (size_t)(h64 + vr) * Tk + j + vch * 8;
    async_b128(g, lds_off(&Vtl[buf][vr][vch * 8]));
  };

  const v16bf q0 = load_frag_a(Qb + h64, 512, qrow0, 0);
  const v16bf q1 = load_frag_a(Qb + h64, 512, qrow0, 32);

  float Mrow[8], Lrow[8];
#pragma unroll
  for (int r = 0; r < 8; ++r) { Mrow[r] = -3.0e38f; Lrow[r] = 0.f; }
  v8f o0 = {}, o1 = {}, o2 = {}, o3 = {};

  issueK(0, 0);
  issueV(0, 0);
  int buf = 0;
  for (int j = 0; j < Tk; j += 32) {
    wait_async0();
    __syncthreads();
    if (j + 32 < Tk) { issueK(buf ^ 1, j + 32); issueV(buf ^ 1, j + 32); }

    const bf16_t* kt = &Kt[buf][0][0];
    v8f s0 = {}, s1 = {};
    s0 = wmma_bf16(q0, load_frag_b(kt, 64, 0, 0), s0);
    s0 = wmma_bf16(q1, load_frag_b(kt, 64, 0, 32), s0);
    s1 = wmma_bf16(q0, load_frag_b(kt, 64, 16, 0), s1);
    s1 = wmma_bf16(q1, load_frag_b(kt, 64, 16, 32), s1);
    float fac[8];
#pragma unroll
    for (int r = 0; r < 8; ++r) {
      const int qm = qrow0 + hi * 8 + r;
      const float* mrow = mask + (size_t)qm * mstride;
      const float a0 = s0[r] + mrow[j + cn];
      const float a1 = s1[r] + mrow[j + 16 + cn];
      float mx = fmaxf(a0, a1);
      mx = fmaxf(mx, __shfl_xor(mx, 1, 32));
      mx = fmaxf(mx, __shfl_xor(mx, 2, 32));
      mx = fmaxf(mx, __shfl_xor(mx, 4, 32));
      mx = fmaxf(mx, __shfl_xor(mx, 8, 32));
      const float nm = fmaxf(Mrow[r], mx);
      fac[r] = __expf(Mrow[r] - nm);
      Mrow[r] = nm;
      const float p0 = __expf(a0 - nm);
      const float p1 = __expf(a1 - nm);
      float ps = p0 + p1;
      ps += __shfl_xor(ps, 1, 32);
      ps += __shfl_xor(ps, 2, 32);
      ps += __shfl_xor(ps, 4, 32);
      ps += __shfl_xor(ps, 8, 32);
      Lrow[r] = Lrow[r] * fac[r] + ps;
      Pt[w][hi * 8 + r][cn] = f2bf(p0);
      Pt[w][hi * 8 + r][16 + cn] = f2bf(p1);
    }
#pragma unroll
    for (int r = 0; r < 8; ++r) {
      o0[r] *= fac[r]; o1[r] *= fac[r]; o2[r] *= fac[r]; o3[r] *= fac[r];
    }
    asm volatile("s_wait_dscnt 0" ::: "memory");  // wave-private Pt, LDS in-order
    const v16bf pa = load_frag_a(&Pt[w][0][0], 32, 0, 0);
    const bf16_t* vt = &Vtl[buf][0][0];
    o0 = wmma_bf16(pa, load_frag_b(vt, 32, 0, 0), o0);
    o1 = wmma_bf16(pa, load_frag_b(vt, 32, 16, 0), o1);
    o2 = wmma_bf16(pa, load_frag_b(vt, 32, 32, 0), o2);
    o3 = wmma_bf16(pa, load_frag_b(vt, 32, 48, 0), o3);
    __syncthreads();
    buf ^= 1;
  }
  v8f os[4] = {o0, o1, o2, o3};
#pragma unroll
  for (int tt = 0; tt < 4; ++tt)
#pragma unroll
    for (int r = 0; r < 8; ++r) {
      const int qm = qrow0 + hi * 8 + r;
      Ob[(size_t)qm * 512 + h64 + tt * 16 + cn] = f2bf(os[tt][r] / Lrow[r]);
    }
}

// out = LN(relu(h + a)) * g + b ; one wave per row of 512. Writes f32 + bf16.
__global__ void __launch_bounds__(256)
add_relu_ln_kernel(const float* __restrict__ hbuf, const float* __restrict__ abuf,
                   const float* __restrict__ g, const float* __restrict__ bb,
                   float* __restrict__ outf, bf16_t* __restrict__ outb) {
  const int w = threadIdx.x >> 5;
  const int lane = threadIdx.x & 31;
  const int row = blockIdx.x * 8 + w;
  const float* hr = hbuf + (size_t)row * 512;
  const float* ar = abuf + (size_t)row * 512;
  float v[16];
  float s = 0.f;
#pragma unroll
  for (int i = 0; i < 16; ++i) {
    const int e = lane + 32 * i;
    v[i] = fmaxf(hr[e] + ar[e], 0.f);
    s += v[i];
  }
  s += __shfl_xor(s, 1, 32); s += __shfl_xor(s, 2, 32); s += __shfl_xor(s, 4, 32);
  s += __shfl_xor(s, 8, 32); s += __shfl_xor(s, 16, 32);
  const float mean = s * (1.f / 512.f);
  float q = 0.f;
#pragma unroll
  for (int i = 0; i < 16; ++i) { const float d = v[i] - mean; q += d * d; }
  q += __shfl_xor(q, 1, 32); q += __shfl_xor(q, 2, 32); q += __shfl_xor(q, 4, 32);
  q += __shfl_xor(q, 8, 32); q += __shfl_xor(q, 16, 32);
  const float inv = rsqrtf(q * (1.f / 512.f) + 1e-5f);
#pragma unroll
  for (int i = 0; i < 16; ++i) {
    const int e = lane + 32 * i;
    const float o = (v[i] - mean) * inv * g[e] + bb[e];
    outf[(size_t)row * 512 + e] = o;
    if (outb) outb[(size_t)row * 512 + e] = f2bf(o);
  }
}

__global__ void cvt_f32_bf16_kernel(const float* __restrict__ in,
                                    bf16_t* __restrict__ out, int n) {
  const int i = blockIdx.x * blockDim.x + threadIdx.x;
  if (i < n) out[i] = f2bf(in[i]);
}
__global__ void copy_f32_kernel(const float* __restrict__ in,
                                float* __restrict__ out, int n) {
  const int i = blockIdx.x * blockDim.x + threadIdx.x;
  if (i < n) out[i] = in[i];
}
// in f32 [R][C] -> out bf16 [C][R]
__global__ void cvtT_f32_bf16_kernel(const float* __restrict__ in,
                                     bf16_t* __restrict__ out, int R, int C) {
  const int i = blockIdx.x * blockDim.x + threadIdx.x;
  if (i < R * C) {
    const int r = i / C, c = i % C;
    out[(size_t)c * R + r] = f2bf(in[i]);
  }
}
// in bf16 [R][C] -> out bf16 [C][R]
__global__ void transpose_bf16_kernel(const bf16_t* __restrict__ in,
                                      bf16_t* __restrict__ out, int R, int C) {
  const int i = blockIdx.x * blockDim.x + threadIdx.x;
  if (i < R * C) {
    const int r = i / C, c = i % C;
    out[(size_t)c * R + r] = in[i];
  }
}

extern "C" void kernel_launch(void* const* d_in, const int* in_sizes, int n_in,
                              void* d_out, int out_size, void* d_ws, size_t ws_size,
                              hipStream_t stream) {
  (void)in_sizes; (void)n_in; (void)out_size; (void)ws_size;
  constexpr int T2 = 2048, T1 = 2048, E = 512, L = 4, F = 2048, H = 8;
  const float* x    = (const float*)d_in[0];
  const float* enc  = (const float*)d_in[1];
  const float* am   = (const float*)d_in[2];
  const float* em   = (const float*)d_in[3];
  const float* k_w  = (const float*)d_in[4];
  const float* k_b  = (const float*)d_in[5];
  const float* v_w  = (const float*)d_in[6];
  const float* v_b  = (const float*)d_in[7];
  const float* q_w  = (const float*)d_in[8];   const float* q_b  = (const float*)d_in[9];
  const float* sk_w = (const float*)d_in[10];  const float* sk_b = (const float*)d_in[11];
  const float* sv_w = (const float*)d_in[12];  const float* sv_b = (const float*)d_in[13];
  const float* o_w  = (const float*)d_in[14];  const float* o_b  = (const float*)d_in[15];
  const float* g1   = (const float*)d_in[16];  const float* b1   = (const float*)d_in[17];
  const float* co_w = (const float*)d_in[18];  const float* co_b = (const float*)d_in[19];
  const float* g2   = (const float*)d_in[20];  const float* b2   = (const float*)d_in[21];
  const float* f1w  = (const float*)d_in[22];  const float* f1b  = (const float*)d_in[23];
  const float* f2w  = (const float*)d_in[24];  const float* f2b  = (const float*)d_in[25];
  const float* g3   = (const float*)d_in[26];  const float* b3   = (const float*)d_in[27];

  char* ws = (char*)d_ws;
  size_t off = 0;
  auto alloc = [&](size_t bytes) {
    void* p = ws + off;
    off = (off + bytes + 255) & ~(size_t)255;
    return p;
  };
  float*  hF   = (float*) alloc((size_t)T2 * E * 4);  // current hidden, f32
  bf16_t* hB   = (bf16_t*)alloc((size_t)T2 * E * 2);  // current hidden, bf16
  bf16_t* encB = (bf16_t*)alloc((size_t)T1 * E * 2);
  bf16_t* KsB  = (bf16_t*)alloc((size_t)T1 * E * 2);  // cross K
  bf16_t* VcT  = (bf16_t*)alloc((size_t)T1 * E * 2);  // cross V^T [E][T1]
  bf16_t* wt   = (bf16_t*)alloc((size_t)E * F * 2);   // reused transposed weight
  bf16_t* qB   = (bf16_t*)alloc((size_t)T2 * E * 2);
  bf16_t* kB   = (bf16_t*)alloc((size_t)T2 * E * 2);
  bf16_t* vB   = (bf16_t*)alloc((size_t)T2 * E * 2);
  bf16_t* vT   = (bf16_t*)alloc((size_t)T2 * E * 2);  // self V^T [E][T2]
  bf16_t* atB  = (bf16_t*)alloc((size_t)T2 * E * 2);  // attention output
  float*  pF   = (float*) alloc((size_t)T2 * E * 4);  // projection output f32
  bf16_t* midB = (bf16_t*)alloc((size_t)T2 * F * 2);  // FFN hidden

  auto cvt = [&](const float* in, bf16_t* out, int n) {
    cvt_f32_bf16_kernel<<<(n + 255) / 256, 256, 0, stream>>>(in, out, n);
  };
  auto cvtT = [&](const float* in, bf16_t* out, int R, int C) {
    cvtT_f32_bf16_kernel<<<(R * C + 255) / 256, 256, 0, stream>>>(in, out, R, C);
  };
  auto trans = [&](const bf16_t* in, bf16_t* out, int R, int C) {
    transpose_bf16_kernel<<<(R * C + 255) / 256, 256, 0, stream>>>(in, out, R, C);
  };
  auto gemm = [&](const bf16_t* A, const bf16_t* W, const float* bias, float* Cf,
                  bf16_t* Cb, int M, int N, int K, int relu) {
    gemm_bf16_kernel<<<dim3(M / 128, N / 64), 256, 0, stream>>>(A, W, bias, Cf, Cb,
                                                                M, N, K, relu);
  };
  auto attn = [&](const bf16_t* Q, const bf16_t* Kk, const bf16_t* Vt,
                  const float* mask, int mstride, bf16_t* O, int Tq, int Tk) {
    attn_kernel<<<dim3(Tq / 128, H), 256, 0, stream>>>(Q, Kk, Vt, mask, mstride, O,
                                                       Tq, Tk);
  };
  auto ln = [&](const float* a, const float* g, const float* b, float* outf,
                bf16_t* outb) {
    add_relu_ln_kernel<<<T2 / 8, 256, 0, stream>>>(hF, a, g, b, outf, outb);
  };

  copy_f32_kernel<<<(T2 * E + 255) / 256, 256, 0, stream>>>(x, hF, T2 * E);
  cvt(x, hB, T2 * E);
  cvt(enc, encB, T1 * E);
  // encoder K/V (computed once)
  cvtT(k_w, wt, E, E);
  gemm(encB, wt, k_b, nullptr, KsB, T1, E, E, 0);
  cvtT(v_w, wt, E, E);
  gemm(encB, wt, v_b, nullptr, vB, T1, E, E, 0);
  trans(vB, VcT, T1, E);

  for (int l = 0; l < L; ++l) {
    const size_t wE = (size_t)l * E * E;
    // ---- self-attention ----
    cvtT(q_w + wE, wt, E, E);
    gemm(hB, wt, q_b + (size_t)l * E, nullptr, qB, T2, E, E, 0);
    cvtT(sk_w + wE, wt, E, E);
    gemm(hB, wt, sk_b + (size_t)l * E, nullptr, kB, T2, E, E, 0);
    cvtT(sv_w + wE, wt, E, E);
    gemm(hB, wt, sv_b + (size_t)l * E, nullptr, vB, T2, E, E, 0);
    trans(vB, vT, T2, E);
    attn(qB, kB, vT, am, T2, atB, T2, T2);
    cvtT(o_w + wE, wt, E, E);
    gemm(atB, wt, o_b + (size_t)l * E, pF, nullptr, T2, E, E, 1);
    ln(pF, g1 + (size_t)l * E, b1 + (size_t)l * E, hF, hB);
    // ---- cross-attention (query = h directly, per reference) ----
    attn(hB, KsB, VcT, em, 0, atB, T2, T1);
    cvtT(co_w + wE, wt, E, E);
    gemm(atB, wt, co_b + (size_t)l * E, pF, nullptr, T2, E, E, 1);
    ln(pF, g2 + (size_t)l * E, b2 + (size_t)l * E, hF, hB);
    // ---- FFN ----
    cvtT(f1w + (size_t)l * E * F, wt, E, F);
    gemm(hB, wt, f1b + (size_t)l * F, nullptr, midB, T2, F, E, 1);
    cvtT(f2w + (size_t)l * F * E, wt, F, E);
    gemm(midB, wt, f2b + (size_t)l * E, pF, nullptr, T2, E, F, 0);
    const bool last = (l == L - 1);
    ln(pF, g3 + (size_t)l * E, b3 + (size_t)l * E,
       last ? (float*)d_out : hF, last ? nullptr : hB);
  }
}
